// RotaryMultiHeadAttention_42460046688623
// MI455X (gfx1250) — matrix operand order repaired and checked by reading, not compiled, not run on hardware
//
#include <hip/hip_runtime.h>
#include <hip/hip_bf16.h>

typedef __attribute__((ext_vector_type(16))) __bf16 v16bf;
typedef __attribute__((ext_vector_type(4)))  __bf16 v4bf;
typedef __attribute__((ext_vector_type(8)))  float  v8f;
typedef unsigned int u32x4 __attribute__((ext_vector_type(4)));
typedef int          i32x4 __attribute__((ext_vector_type(4)));
typedef int          i32x8 __attribute__((ext_vector_type(8)));

union FragBF { uint4 q[2]; v16bf v; };

#define WMMA_BF16(a, b, c) \
  __builtin_amdgcn_wmma_f32_16x16x32_bf16(false, (a), false, (b), (short)0, (c), false, false)

static constexpr int B_  = 4;
static constexpr int T_  = 1024;
static constexpr int D_  = 512;
static constexpr int H_  = 8;
static constexpr int DK_ = 64;
static constexpr int P_  = 2047;   // 2*T - 1

enum { EPI_QUV = 0, EPI_HEAD = 1, EPI_VT = 2, EPI_F32 = 3 };

#if defined(__has_builtin)
#if __has_builtin(__builtin_amdgcn_tensor_load_to_lds) && __has_builtin(__builtin_amdgcn_s_wait_tensorcnt)
#define USE_TDM 1
#endif
#endif
#ifndef USE_TDM
#define USE_TDM 0
#endif

// low 32 bits of a generic pointer to a __shared__ object == LDS byte offset
__device__ inline unsigned lds_off(const void* p) {
    return (unsigned)(unsigned long long)(uintptr_t)p;
}

// ---------------------------------------------------------------------------
// 2-D TDM load: tile_d1 rows of tile_d0 bf16 elements, row stride stride_d0
// elements, into contiguous LDS at ldst. D# built per CDNA5 ISA §8.3/8.4.
// ---------------------------------------------------------------------------
__device__ inline void tdm_load_2d(__bf16* ldst, const void* gptr,
                                   unsigned tile_d0, unsigned tile_d1,
                                   unsigned stride_d0,
                                   unsigned tensor_d0, unsigned tensor_d1)
{
#if USE_TDM
    const unsigned long long ga = (unsigned long long)gptr;
    u32x4 g0;
    g0[0] = 1u;                                      // count=1, user descriptor
    g0[1] = lds_off(ldst);                           // lds_addr
    g0[2] = (unsigned)(ga & 0xffffffffu);            // global_addr[31:0]
    g0[3] = (unsigned)((ga >> 32) & 0x1ffffffu)      // global_addr[56:32]
          | (2u << 30);                              // type = 2 ("image")
    i32x8 g1;
    g1[0] = (int)(1u << 16);                         // data_size=1 (2B), mask=0
    g1[1] = (int)((tensor_d0 & 0xffffu) << 16);      // tensor_dim0[15:0]
    g1[2] = (int)((tensor_d0 >> 16) | ((tensor_d1 & 0xffffu) << 16));
    g1[3] = (int)((tensor_d1 >> 16) | (tile_d0 << 16));  // tile_dim0
    g1[4] = (int)(tile_d1 & 0xffffu);                // tile_dim1 (tile_dim2=0)
    g1[5] = (int)stride_d0;                          // tensor_dim0_stride[31:0]
    g1[6] = 0;
    g1[7] = 0;
    const i32x4 z4 = {0, 0, 0, 0};
#if __clang_major__ >= 23
    const i32x8 z8 = {0, 0, 0, 0, 0, 0, 0, 0};
    __builtin_amdgcn_tensor_load_to_lds(g0, g1, z4, z4, z8, 0);
#else
    __builtin_amdgcn_tensor_load_to_lds(g0, g1, z4, z4, 0);
#endif
#else
    // Fallback: cooperative wave32 copy (row-strided), 16B chunks.
    const int lane = threadIdx.x & 31;
    const unsigned chunksPerRow = tile_d0 / 8;       // tile_d0 % 8 == 0 here
    const unsigned nChunks = chunksPerRow * tile_d1;
    for (unsigned i = lane; i < nChunks; i += 32) {
        const unsigned row = i / chunksPerRow;
        const unsigned col = (i - row * chunksPerRow) * 8;
        *(uint4*)(ldst + (unsigned long long)row * tile_d0 + col) =
            *(const uint4*)((const __bf16*)gptr + (unsigned long long)row * stride_d0 + col);
    }
#endif
}

template<int N>
__device__ inline void tdm_wait()
{
#if USE_TDM
    __builtin_amdgcn_s_wait_tensorcnt((short)N);
#endif
    asm volatile("" ::: "memory");
}

// ---------------------------------------------------------------------------
// Generic tiled GEMM: C[M x 512] = A[M x K] @ Bt[512 x K]^T  (Bt row-major NxK)
// 128 threads (4 waves), 64x64 tile, each wave 32x32 via 2x2 WMMA fragments.
// ---------------------------------------------------------------------------
template<bool AF32>
__global__ __launch_bounds__(128)
void gemm_tn_kernel(const void* __restrict__ Ao, long lda,
                    const float* __restrict__ Bt, long ldb,
                    const float* __restrict__ bias,
                    int M, int K,
                    int epi, long rows,
                    void* __restrict__ out0, void* __restrict__ out1,
                    const float* __restrict__ add0, const float* __restrict__ add1,
                    long ldo)
{
    __shared__ __align__(16) __bf16 As[64][40];
    __shared__ __align__(16) __bf16 Bs[64][40];

    const int tid  = threadIdx.x;
    const int lane = tid & 31;
    const int wave = tid >> 5;
    const int wm   = (wave >> 1) * 32;
    const int wn   = (wave & 1) * 32;
    const int m0   = blockIdx.y * 64;
    const int n0   = blockIdx.x * 64;
    const int am   = lane & 15;
    const int ah   = lane >> 4;

    v8f acc[2][2] = {};

    const int fr = tid >> 3;            // 0..15
    const int fc = (tid & 7) * 4;       // 0..28

    for (int kt = 0; kt < K; kt += 32) {
        if (kt + 32 < K) {  // speculative prefetch of next K tiles
            __builtin_prefetch((const float*)Bt + (long)(n0 + fr) * ldb + kt + 32 + fc, 0, 1);
            if constexpr (AF32)
                __builtin_prefetch((const float*)Ao + (long)(m0 + fr) * lda + kt + 32 + fc, 0, 1);
            else
                __builtin_prefetch((const __bf16*)Ao + (long)(m0 + fr) * lda + kt + 32 + fc, 0, 1);
        }
        #pragma unroll
        for (int rr = 0; rr < 64; rr += 16) {
            const int gm = m0 + fr + rr;
            if constexpr (AF32) {
                float4 a4 = make_float4(0.f, 0.f, 0.f, 0.f);
                if (gm < M)
                    a4 = *(const float4*)((const float*)Ao + (long)gm * lda + kt + fc);
                As[fr + rr][fc + 0] = (__bf16)a4.x;
                As[fr + rr][fc + 1] = (__bf16)a4.y;
                As[fr + rr][fc + 2] = (__bf16)a4.z;
                As[fr + rr][fc + 3] = (__bf16)a4.w;
            } else {
                v4bf a4 = {};
                if (gm < M)
                    a4 = *(const v4bf*)((const __bf16*)Ao + (long)gm * lda + kt + fc);
                As[fr + rr][fc + 0] = a4[0];
                As[fr + rr][fc + 1] = a4[1];
                As[fr + rr][fc + 2] = a4[2];
                As[fr + rr][fc + 3] = a4[3];
            }
        }
        #pragma unroll
        for (int rr = 0; rr < 64; rr += 16) {
            const float4 b4 = *(const float4*)(Bt + (long)(n0 + fr + rr) * ldb + kt + fc);
            Bs[fr + rr][fc + 0] = (__bf16)b4.x;
            Bs[fr + rr][fc + 1] = (__bf16)b4.y;
            Bs[fr + rr][fc + 2] = (__bf16)b4.z;
            Bs[fr + rr][fc + 3] = (__bf16)b4.w;
        }
        __syncthreads();

        FragBF fa[2], fb[2];
        #pragma unroll
        for (int i = 0; i < 2; ++i) {
            const __bf16* p = &As[wm + i * 16 + am][ah * 8];
            fa[i].q[0] = *(const uint4*)p;
            fa[i].q[1] = *(const uint4*)(p + 16);
        }
        #pragma unroll
        for (int j = 0; j < 2; ++j) {
            const __bf16* p = &Bs[wn + j * 16 + am][ah * 8];
            fb[j].q[0] = *(const uint4*)p;
            fb[j].q[1] = *(const uint4*)(p + 16);
        }
        #pragma unroll
        for (int i = 0; i < 2; ++i)
            #pragma unroll
            for (int j = 0; j < 2; ++j)
                acc[i][j] = WMMA_BF16(fa[i].v, fb[j].v, acc[i][j]);
        __syncthreads();
    }

    #pragma unroll
    for (int i = 0; i < 2; ++i) {
        #pragma unroll
        for (int j = 0; j < 2; ++j) {
            const int nb = n0 + wn + j * 16 + am;
            const float bb = bias ? bias[nb] : 0.f;
            #pragma unroll
            for (int r = 0; r < 8; ++r) {
                const int mm = m0 + wm + i * 16 + r + 8 * ah;
                if (mm >= M) continue;
                const float val = acc[i][j][r] + bb;
                const long bi = mm / rows;
                const long ti = mm - bi * rows;
                if (epi == EPI_QUV) {
                    const long off = ((bi * H_ + (nb >> 6)) * rows + ti) * DK_ + (nb & 63);
                    ((__bf16*)out0)[off] = (__bf16)(val + add0[nb]);
                    ((__bf16*)out1)[off] = (__bf16)(val + add1[nb]);
                } else if (epi == EPI_HEAD) {
                    const long off = ((bi * H_ + (nb >> 6)) * rows + ti) * DK_ + (nb & 63);
                    ((__bf16*)out0)[off] = (__bf16)val;
                } else if (epi == EPI_VT) {
                    const long off = ((bi * H_ + (nb >> 6)) * (long)DK_ + (nb & 63)) * rows + ti;
                    ((__bf16*)out0)[off] = (__bf16)val;
                } else { // EPI_F32
                    ((float*)out0)[(long)mm * ldo + nb] = val;
                }
            }
        }
    }
}

// ---------------------------------------------------------------------------
// Fused relative-position flash attention with TDM-staged K/V tiles.
// One wave = 16 query rows of one (b,h).
// ---------------------------------------------------------------------------
__global__ __launch_bounds__(32)
void relattn_kernel(const __bf16* __restrict__ QU, const __bf16* __restrict__ QV,
                    const __bf16* __restrict__ KH, const __bf16* __restrict__ VT,
                    const __bf16* __restrict__ PH, __bf16* __restrict__ ctx)
{
    __shared__ __align__(16) __bf16 BDs[16][1040];   // 33.3 KB shifted-BD band
    __shared__ __align__(16) __bf16 Pb[16][32];      // C->A layout bounce
    __shared__ __align__(16) __bf16 Ks[2][32][64];   // double-buffered K tile (TDM)
    __shared__ __align__(16) __bf16 Vs[2][64][32];   // double-buffered V tile (TDM)

    const int lane = threadIdx.x;
    const int am = lane & 15, ah = lane >> 4;
    const int q0 = blockIdx.x * 16;
    const int h  = blockIdx.y;
    const int b  = blockIdx.z;
    const long bh = (long)b * H_ + h;
    const __bf16* QUb = QU + (bh * T_ + q0) * DK_;
    const __bf16* QVb = QV + (bh * T_ + q0) * DK_;
    const __bf16* KHb = KH + bh * T_ * DK_;
    const __bf16* VTb = VT + bh * (long)DK_ * T_;
    const __bf16* PHb = PH + (long)h * P_ * DK_;

    // Q fragments (A operand, 16x32, two K-steps covering dk=64)
    FragBF qu[2], qv[2];
    #pragma unroll
    for (int ks = 0; ks < 2; ++ks) {
        const __bf16* p = QUb + am * DK_ + ks * 32 + ah * 8;
        qu[ks].q[0] = *(const uint4*)p;
        qu[ks].q[1] = *(const uint4*)(p + 16);
        p = QVb + am * DK_ + ks * 32 + ah * 8;
        qv[ks].q[0] = *(const uint4*)p;
        qv[ks].q[1] = *(const uint4*)(p + 16);
    }

    // kick off TDM for the first K/V tiles (kt = 0, buffer 0)
    tdm_load_2d(&Ks[0][0][0], KHb, DK_, 32, DK_, DK_, 1u << 20);
    tdm_load_2d(&Vs[0][0][0], VTb, 32, DK_, T_, 32, DK_);

    // ---- Phase 1: BD band over p in [pbase, pbase + 1040) ----
    const int pbase = (T_ - 16) - q0;   // 1008 - q0 >= 0
    for (int pt = 0; pt < 65; ++pt) {
        v8f c = {};
        const int pr = pbase + pt * 16 + am;   // p row (B-operand N index)
        __builtin_prefetch(PHb + (long)(pr + 32) * DK_, 0, 1);  // speculative
        #pragma unroll
        for (int ks = 0; ks < 2; ++ks) {
            FragBF fb;
            if (pr < P_) {
                const __bf16* p = PHb + (long)pr * DK_ + ks * 32 + ah * 8;
                fb.q[0] = *(const uint4*)p;
                fb.q[1] = *(const uint4*)(p + 16);
            } else {
                fb.q[0] = make_uint4(0u, 0u, 0u, 0u);
                fb.q[1] = make_uint4(0u, 0u, 0u, 0u);
            }
            c = WMMA_BF16(qv[ks].v, fb.v, c);
        }
        const int col = pt * 16 + am;
        #pragma unroll
        for (int r = 0; r < 8; ++r)
            BDs[r + 8 * ah][col] = (__bf16)c[r];
    }
    __syncthreads();

    // ---- Phase 2: flash loop over K/V in steps of 32 ----
    float mrow[8], lrow[8];
    #pragma unroll
    for (int r = 0; r < 8; ++r) { mrow[r] = -1e30f; lrow[r] = 0.f; }
    v8f o[4] = {};

    for (int kt = 0; kt < T_; kt += 32) {
        const int buf = (kt >> 5) & 1;
        if (kt + 32 < T_) {
            // async-stage next tile while computing this one
            tdm_load_2d(&Ks[buf ^ 1][0][0], KHb + (long)(kt + 32) * DK_,
                        DK_, 32, DK_, DK_, 1u << 20);
            tdm_load_2d(&Vs[buf ^ 1][0][0], VTb + (kt + 32),
                        32, DK_, T_, 32, DK_);
            tdm_wait<2>();   // current tile's pair complete, next pair in flight
        } else {
            tdm_wait<0>();
        }

        v8f s[2];
        #pragma unroll
        for (int t2 = 0; t2 < 2; ++t2) {
            const int n0 = kt + t2 * 16;
            v8f c = {};
            #pragma unroll
            for (int ks = 0; ks < 2; ++ks) {
                FragBF fb;
                const __bf16* p = &Ks[buf][t2 * 16 + am][ks * 32 + ah * 8];
                fb.q[0] = *(const uint4*)p;
                fb.q[1] = *(const uint4*)(p + 16);
                c = WMMA_BF16(qu[ks].v, fb.v, c);
            }
            #pragma unroll
            for (int r = 0; r < 8; ++r) {
                const int m = r + 8 * ah;
                // shifted BD: raw col (n + 1023 - q) - pbase == n + 15 - m
                c[r] = (c[r] + (float)BDs[m][n0 + am + 15 - m]) * 0.125f;
            }
            s[t2] = c;
        }
        // online softmax (row m spans one 16-lane half per VGPR r)
        #pragma unroll
        for (int r = 0; r < 8; ++r) {
            float mx = fmaxf(s[0][r], s[1][r]);
            #pragma unroll
            for (int off = 1; off < 16; off <<= 1)
                mx = fmaxf(mx, __shfl_xor(mx, off, 32));
            const float mnew = fmaxf(mrow[r], mx);
            const float corr = __expf(mrow[r] - mnew);
            mrow[r] = mnew;
            const float p0 = __expf(s[0][r] - mnew);
            const float p1 = __expf(s[1][r] - mnew);
            s[0][r] = p0; s[1][r] = p1;
            float ps = p0 + p1;
            #pragma unroll
            for (int off = 1; off < 16; off <<= 1)
                ps += __shfl_xor(ps, off, 32);
            lrow[r] = lrow[r] * corr + ps;
            #pragma unroll
            for (int j = 0; j < 4; ++j) o[j][r] *= corr;
        }
        // P tile: C layout -> LDS -> A layout
        #pragma unroll
        for (int t2 = 0; t2 < 2; ++t2)
            #pragma unroll
            for (int r = 0; r < 8; ++r)
                Pb[r + 8 * ah][t2 * 16 + am] = (__bf16)s[t2][r];
        __syncthreads();
        FragBF pa;
        {
            const __bf16* p = &Pb[am][ah * 8];
            pa.q[0] = *(const uint4*)p;
            pa.q[1] = *(const uint4*)(p + 16);
        }
        #pragma unroll
        for (int j = 0; j < 4; ++j) {
            FragBF fv;
            const __bf16* p = &Vs[buf][j * 16 + am][ah * 8];
            fv.q[0] = *(const uint4*)p;
            fv.q[1] = *(const uint4*)(p + 16);
            o[j] = WMMA_BF16(pa.v, fv.v, o[j]);
        }
        __syncthreads();
    }

    // normalize and write ctx [b*T + q][h*64 + dk] as bf16
    #pragma unroll
    for (int j = 0; j < 4; ++j)
        #pragma unroll
        for (int r = 0; r < 8; ++r) {
            const int m = r + 8 * ah;
            const float val = o[j][r] / lrow[r];
            ctx[((long)b * T_ + q0 + m) * D_ + h * DK_ + j * 16 + am] = (__bf16)val;
        }
}

// ---------------------------------------------------------------------------
extern "C" void kernel_launch(void* const* d_in, const int* in_sizes, int n_in,
                              void* d_out, int out_size, void* d_ws, size_t ws_size,
                              hipStream_t stream)
{
    const float* q   = (const float*)d_in[0];
    const float* k   = (const float*)d_in[1];
    const float* v   = (const float*)d_in[2];
    const float* pos = (const float*)d_in[3];
    const float* Wq  = (const float*)d_in[4];
    const float* bq  = (const float*)d_in[5];
    const float* Wk  = (const float*)d_in[6];
    const float* bk  = (const float*)d_in[7];
    const float* Wv  = (const float*)d_in[8];
    const float* bv  = (const float*)d_in[9];
    const float* Wp  = (const float*)d_in[10];
    const float* pbu = (const float*)d_in[11];   // (H,DK) flat = 512, indexed by n
    const float* pbv = (const float*)d_in[12];
    const float* Wo  = (const float*)d_in[13];
    const float* bo  = (const float*)d_in[14];

    char* w = (char*)d_ws;
    const size_t S1 = (size_t)B_ * H_ * T_ * DK_;   // 2,097,152 elems
    const size_t SP = (size_t)H_ * P_ * DK_;        // 1,048,064 elems
    __bf16* QU  = (__bf16*)w; w += S1 * 2;
    __bf16* QV  = (__bf16*)w; w += S1 * 2;
    __bf16* KH  = (__bf16*)w; w += S1 * 2;
    __bf16* VT  = (__bf16*)w; w += S1 * 2;
    __bf16* PH  = (__bf16*)w; w += ((SP * 2 + 255) / 256) * 256;
    __bf16* CTX = (__bf16*)w;                       // total ~22 MB of ws

    const dim3 blk(128);
    const dim3 g512(D_ / 64, (B_ * T_) / 64);       // 8 x 64 tiles
    gemm_tn_kernel<true><<<g512, blk, 0, stream>>>(
        q, D_, Wq, D_, bq, B_ * T_, D_, EPI_QUV, T_, QU, QV, pbu, pbv, 0);
    gemm_tn_kernel<true><<<g512, blk, 0, stream>>>(
        k, D_, Wk, D_, bk, B_ * T_, D_, EPI_HEAD, T_, KH, nullptr, nullptr, nullptr, 0);
    gemm_tn_kernel<true><<<g512, blk, 0, stream>>>(
        v, D_, Wv, D_, bv, B_ * T_, D_, EPI_VT, T_, VT, nullptr, nullptr, nullptr, 0);
    const dim3 gp(D_ / 64, (P_ + 63) / 64);         // 8 x 32 tiles, M=2047 guarded
    gemm_tn_kernel<true><<<gp, blk, 0, stream>>>(
        pos, D_, Wp, D_, nullptr, P_, D_, EPI_HEAD, P_, PH, nullptr, nullptr, nullptr, 0);

    const dim3 ga(T_ / 16, H_, B_);                 // 2048 waves
    relattn_kernel<<<ga, dim3(32), 0, stream>>>(QU, QV, KH, VT, PH, CTX);

    gemm_tn_kernel<false><<<g512, blk, 0, stream>>>(
        CTX, D_, Wo, D_, bo, B_ * T_, D_, EPI_F32, T_, d_out, nullptr, nullptr, nullptr, D_);
}
